// CausalSelfAttention_26396869001359
// MI455X (gfx1250) — matrix operand order repaired and checked by reading, not compiled, not run on hardware
//
#include <hip/hip_runtime.h>

typedef __attribute__((ext_vector_type(16))) _Float16 v16h;
typedef __attribute__((ext_vector_type(8)))  _Float16 v8h;
typedef __attribute__((ext_vector_type(8)))  float    v8f;
typedef __attribute__((ext_vector_type(4)))  unsigned int v4u;
typedef __attribute__((ext_vector_type(4)))  int v4i;
typedef __attribute__((ext_vector_type(8)))  int v8i;

union H16 { v16h v; v8h h[2]; };

#define B_ 4
#define T_ 1024
#define C_ 2048
#define H_ 16
#define D_ 128

#if defined(__has_builtin)
#if __has_builtin(__builtin_amdgcn_tensor_load_to_lds)
#define HAVE_TDM 1
#endif
#endif

static __device__ __forceinline__ v8f wmma_f16(const H16& a, const H16& b, v8f c) {
    // D = A(16x32 f16) * B(32x16 f16) + C(16x16 f32)
    return __builtin_amdgcn_wmma_f32_16x16x32_f16(
        false, a.v, false, b.v, (short)0, c, false, false);
}

// LDS byte offset of a shared-memory object (flat->LDS truncation per ISA 10.2)
static __device__ __forceinline__ unsigned lds_off(const void* p) {
    return (unsigned)(unsigned long long)p;
}
// Async DMA: 16B global -> LDS, tracked by ASYNCcnt (gfx1250)
static __device__ __forceinline__ void async_b128(const void* lds, const void* g) {
    asm volatile("global_load_async_to_lds_b128 %0, %1, off"
                 :: "v"(lds_off(lds)), "v"((unsigned long long)g) : "memory");
}
static __device__ __forceinline__ void wait_async0() {
    asm volatile("s_wait_asynccnt 0x0" ::: "memory");
}

// ---------------------------------------------------------------------------
// fp32 -> f16 conversion (bandwidth bound, grid-stride)
// ---------------------------------------------------------------------------
__global__ void f32_to_f16_kernel(const float* __restrict__ in,
                                  _Float16* __restrict__ out, size_t n) {
    size_t i = (size_t)blockIdx.x * blockDim.x + threadIdx.x;
    size_t stride = (size_t)gridDim.x * blockDim.x;
    for (; i < n; i += stride) out[i] = (_Float16)in[i];
}

// ---------------------------------------------------------------------------
// In-place RoPE on [B*H, T, D] f16; pairs (d, d+64); scale folds 1/sqrt(D) into q
// ---------------------------------------------------------------------------
__global__ void rope_kernel(_Float16* __restrict__ X, float scale) {
    const size_t total = (size_t)B_ * H_ * T_ * (D_ / 2);
    size_t idx = (size_t)blockIdx.x * blockDim.x + threadIdx.x;
    if (idx >= total) return;
    int d2 = (int)(idx & 63);
    int t  = (int)((idx >> 6) & (T_ - 1));
    size_t bh = idx >> 16;
    float inv_freq = __powf(10000.0f, -(float)(2 * d2) * (1.0f / (float)D_));
    float ang = (float)t * inv_freq;
    float s, c;
    __sincosf(ang, &s, &c);
    size_t base = (bh * T_ + t) * D_ + d2;
    float x1 = (float)X[base];
    float x2 = (float)X[base + 64];
    X[base]      = (_Float16)((x1 * c - x2 * s) * scale);
    X[base + 64] = (_Float16)((x1 * s + x2 * c) * scale);
}

// ---------------------------------------------------------------------------
// f16 WMMA GEMM: C[M,N] = A[M,K] * B[K,N] + bias[N]
// Block tile 128x128, 8 waves (2M x 4N), wave tile 64x32, K-tile 64 (2 WMMA steps).
// A tile staged with global_load_async_to_lds_b128 (ASYNCcnt); B tile staged
// synchronously (transpose scatter). One barrier per K-tile.
// EPI==0: f32 out.  EPI==1: qkv scatter into Qh/Kh/Vh [B,H,T,D] f16.
// ---------------------------------------------------------------------------
template <int EPI>
__global__ __launch_bounds__(256)
void gemm_kernel(const _Float16* __restrict__ A, const _Float16* __restrict__ Bw,
                 const float* __restrict__ bias, float* __restrict__ outF,
                 _Float16* __restrict__ Qh, _Float16* __restrict__ Kh,
                 _Float16* __restrict__ Vh, int M, int N, int K) {
    __shared__ __align__(16) _Float16 As[2][128][72];  // [m][k], 144B rows
    __shared__ __align__(16) _Float16 Bt[2][128][72];  // transposed [n][k]

    const int tid  = threadIdx.x;
    const int lane = tid & 31;
    const int wave = tid >> 5;
    const int wm   = wave >> 2;        // 0..1
    const int wn   = wave & 3;         // 0..3
    const int hi   = (lane >> 4) & 1;
    const int l16  = lane & 15;
    const int m0   = blockIdx.y * 128;
    const int n0   = blockIdx.x * 128;

    // cooperative-load coordinates (K-tile = 64 halves)
    const int ar = tid >> 1;           // A row 0..127
    const int ac = (tid & 1) * 32;     // A col 0 or 32
    const int bk = tid >> 2;           // B k 0..63
    const int bn = (tid & 3) * 32;     // B n 0,32,64,96

    v8f zero = {};
    v8f acc[4][2];
#pragma unroll
    for (int i = 0; i < 4; i++) {
#pragma unroll
        for (int j = 0; j < 2; j++) acc[i][j] = zero;
    }

    const int nk = K >> 6;

    // ---- preload tile 0 ----
    {
        const _Float16* ga = A + (size_t)(m0 + ar) * K + ac;
#pragma unroll
        for (int q = 0; q < 4; q++) async_b128(&As[0][ar][ac + q * 8], ga + q * 8);
        const v8h* pb = (const v8h*)(Bw + (size_t)bk * N + n0 + bn);
        v8h b0 = pb[0], b1 = pb[1], b2 = pb[2], b3 = pb[3];
#pragma unroll
        for (int j = 0; j < 8; j++) {
            Bt[0][bn + j][bk]      = b0[j];
            Bt[0][bn + 8 + j][bk]  = b1[j];
            Bt[0][bn + 16 + j][bk] = b2[j];
            Bt[0][bn + 24 + j][bk] = b3[j];
        }
        wait_async0();
    }
    __syncthreads();

    for (int kt = 0; kt < nk; ++kt) {
        const int s = kt & 1;
        const bool more = (kt + 1) < nk;
        v8h nb0, nb1, nb2, nb3;
        if (more) {
            const int kn = (kt + 1) << 6;
            const _Float16* ga = A + (size_t)(m0 + ar) * K + kn + ac;
#pragma unroll
            for (int q = 0; q < 4; q++) async_b128(&As[s ^ 1][ar][ac + q * 8], ga + q * 8);
            const v8h* pb = (const v8h*)(Bw + (size_t)(kn + bk) * N + n0 + bn);
            nb0 = pb[0]; nb1 = pb[1]; nb2 = pb[2]; nb3 = pb[3];
            __builtin_prefetch(A + (size_t)(m0 + ar) * K + ((kt + 2) << 6) + ac, 0, 1);
            __builtin_prefetch(Bw + (size_t)(((kt + 2) << 6) + bk) * N + n0 + bn, 0, 1);
        }

        // ---- compute: 2 WMMA K-steps x 4x2 tiles ----
#pragma unroll
        for (int ks = 0; ks < 2; ks++) {
            const int ko = ks * 32;
            H16 af[4];
#pragma unroll
            for (int i = 0; i < 4; i++) {
                const int r  = wm * 64 + i * 16 + l16;   // A: lane = row
                const int kb = ko + hi * 8;              // halves K in {kb.., 16+kb..}
                af[i].h[0] = *(const v8h*)&As[s][r][kb];
                af[i].h[1] = *(const v8h*)&As[s][r][kb + 16];
            }
            H16 bf[2];
#pragma unroll
            for (int j = 0; j < 2; j++) {
                const int n  = wn * 32 + j * 16 + l16;   // B: lane = col
                const int kc = ko + hi * 8;              // halves K in {kc.., 16+kc..}
                bf[j].h[0] = *(const v8h*)&Bt[s][n][kc];
                bf[j].h[1] = *(const v8h*)&Bt[s][n][kc + 16];
            }
#pragma unroll
            for (int i = 0; i < 4; i++) {
#pragma unroll
                for (int j = 0; j < 2; j++) acc[i][j] = wmma_f16(af[i], bf[j], acc[i][j]);
            }
        }

        if (more) {
            const int s2 = s ^ 1;
#pragma unroll
            for (int j = 0; j < 8; j++) {
                Bt[s2][bn + j][bk]      = nb0[j];
                Bt[s2][bn + 8 + j][bk]  = nb1[j];
                Bt[s2][bn + 16 + j][bk] = nb2[j];
                Bt[s2][bn + 24 + j][bk] = nb3[j];
            }
            wait_async0();   // this wave's async A-tile writes have landed
        }
        __syncthreads();
    }

    // ---- epilogue: C layout VGPR g -> M = g + 8*hi, N = l16 ----
#pragma unroll
    for (int i = 0; i < 4; i++) {
#pragma unroll
        for (int j = 0; j < 2; j++) {
#pragma unroll
            for (int g = 0; g < 8; g++) {
                const int row = m0 + wm * 64 + i * 16 + g + hi * 8;
                const int col = n0 + wn * 32 + j * 16 + l16;
                const float v = acc[i][j][g] + bias[col];
                if (EPI == 0) {
                    outF[(size_t)row * N + col] = v;
                } else {
                    const int which = col >> 11;      // 0=q 1=k 2=v
                    const int rem   = col & 2047;
                    const int h     = rem >> 7;
                    const int d     = rem & 127;
                    const int b     = row >> 10;
                    const int t     = row & 1023;
                    _Float16* dst = (which == 0) ? Qh : ((which == 1) ? Kh : Vh);
                    dst[(((size_t)(b * H_ + h)) * T_ + t) * D_ + d] = (_Float16)v;
                }
            }
        }
    }
}

// ---------------------------------------------------------------------------
// Flash attention: one block per (b, h, 128-row q tile). 8 waves x 16 q rows.
// K tile staged by the Tensor Data Mover (TENSORcnt) with LDS padding;
// V transposed via lane scatter; causal online softmax; WMMA everywhere.
// ---------------------------------------------------------------------------
__global__ __launch_bounds__(256)
void flash_kernel(const _Float16* __restrict__ Qh, const _Float16* __restrict__ Kh,
                  const _Float16* __restrict__ Vh, _Float16* __restrict__ Yh) {
    __shared__ __align__(16) _Float16 Ksm[64][136];    // [key][d], 272B rows (TDM pad)
    __shared__ __align__(16) _Float16 Vt[128][72];     // [d][key]
    __shared__ __align__(16) _Float16 Psm[8][16][72];  // per-wave P staging

    const int tid  = threadIdx.x;
    const int lane = tid & 31;
    const int wave = tid >> 5;
    const int hi   = (lane >> 4) & 1;
    const int l16  = lane & 15;
    const int qt   = blockIdx.x & 7;
    const int bh   = blockIdx.x >> 3;

    const _Float16* Qb = Qh + (size_t)bh * T_ * D_;
    const _Float16* Kb = Kh + (size_t)bh * T_ * D_;
    const _Float16* Vb = Vh + (size_t)bh * T_ * D_;

    const int qr0   = qt * 128 + wave * 16;
    const int myrow = qr0 + l16;

    H16 qa[4];
#pragma unroll
    for (int c = 0; c < 4; c++) {
        const _Float16* p = Qb + (size_t)myrow * D_ + c * 32 + hi * 8;
        qa[c].h[0] = *(const v8h*)p;
        qa[c].h[1] = *(const v8h*)(p + 16);
    }

    v8f zero = {};
    v8f O[8];
    float m[8], l[8];
#pragma unroll
    for (int n = 0; n < 8; n++) O[n] = zero;
#pragma unroll
    for (int g = 0; g < 8; g++) { m[g] = -1e30f; l[g] = 0.0f; }

    const int nkb = qt * 2 + 2;
    const int lr  = tid >> 2;          // 0..63
    const int lc  = (tid & 3) * 32;

    for (int kb = 0; kb < nkb; ++kb) {
        const int k0 = kb * 64;
#ifdef HAVE_TDM
        if (wave == 0) {
            // D# group0: count=1 | lds_addr | 57-bit global addr | type=2
            const unsigned long long ga =
                (unsigned long long)(const void*)(Kb + (size_t)k0 * D_);
            v4u g0;
            g0[0] = 1u;
            g0[1] = lds_off(&Ksm[0][0]);
            g0[2] = (unsigned)ga;
            g0[3] = ((unsigned)(ga >> 32) & 0x01FFFFFFu) | (2u << 30);
            // D# group1: data_size=2B, pad_enable, pad_interval=64 DW, pad_amount=4 DW
            // tensor_dim0=128, tensor_dim1=64, tile_dim0=128, tile_dim1=64, stride0=128
            v8i g1;
            g1[0] = (int)((1u << 16) | (1u << 20) | (5u << 22) | (3u << 25));
            g1[1] = (int)(128u << 16);   // tensor_dim0 lo16
            g1[2] = (int)(64u << 16);    // tensor_dim0 hi16=0 | tensor_dim1 lo16
            g1[3] = (int)(128u << 16);   // tensor_dim1 hi16=0 | tile_dim0
            g1[4] = (int)64u;            // tile_dim1 | tile_dim2=0
            g1[5] = (int)128u;           // tensor_dim0_stride lo32
            g1[6] = 0;                   // stride0 hi16 | stride1 lo16
            g1[7] = 0;
            v4i z4 = { 0, 0, 0, 0 };
            v8i z8 = { 0, 0, 0, 0, 0, 0, 0, 0 };
            __builtin_amdgcn_tensor_load_to_lds(g0, g1, z4, z4, z8, 0);
            __builtin_amdgcn_s_wait_tensorcnt(0);
        }
#else
        {
            const v8h* ks = (const v8h*)(Kb + (size_t)(k0 + lr) * D_ + lc);
            v8h k0v = ks[0], k1v = ks[1], k2v = ks[2], k3v = ks[3];
            v8h* kd = (v8h*)&Ksm[lr][lc];
            kd[0] = k0v; kd[1] = k1v; kd[2] = k2v; kd[3] = k3v;
        }
#endif
        // ---- stage V transposed ----
        {
            const v8h* vs = (const v8h*)(Vb + (size_t)(k0 + lr) * D_ + lc);
            v8h vv[4] = { vs[0], vs[1], vs[2], vs[3] };
#pragma unroll
            for (int q = 0; q < 4; q++) {
#pragma unroll
                for (int e = 0; e < 8; e++) Vt[lc + q * 8 + e][lr] = vv[q][e];
            }
        }
        __syncthreads();

        // ---- S = q . k^T over 4 key sub-tiles of 16 ----
        v8f S[4];
#pragma unroll
        for (int kt = 0; kt < 4; kt++) {
            v8f s = zero;
#pragma unroll
            for (int c = 0; c < 4; c++) {
                H16 bf;  // B = K^T: lane = key col, halves = d {8hi.., 16+8hi..}
                const _Float16* kp = &Ksm[kt * 16 + l16][c * 32 + hi * 8];
                bf.h[0] = *(const v8h*)kp;
                bf.h[1] = *(const v8h*)(kp + 16);
                s = wmma_f16(qa[c], bf, s);
            }
            const int key = k0 + kt * 16 + l16;
#pragma unroll
            for (int g = 0; g < 8; g++) {
                const int rowg = qr0 + g + hi * 8;
                if (key > rowg) s[g] = -1e30f;   // causal mask (uniform EXEC)
            }
            S[kt] = s;
        }

        // ---- online softmax per q row (16-lane reductions, wave32) ----
#pragma unroll
        for (int g = 0; g < 8; g++) {
            float mx = fmaxf(fmaxf(S[0][g], S[1][g]), fmaxf(S[2][g], S[3][g]));
#pragma unroll
            for (int off = 1; off < 16; off <<= 1)
                mx = fmaxf(mx, __shfl_xor(mx, off, 32));
            const float mn    = fmaxf(m[g], mx);
            const float alpha = __expf(m[g] - mn);
            m[g] = mn;
            float psum = 0.0f;
#pragma unroll
            for (int kt = 0; kt < 4; kt++) {
                const float p = __expf(S[kt][g] - mn);
                psum += p;
                Psm[wave][g + hi * 8][kt * 16 + l16] = (_Float16)p;
            }
#pragma unroll
            for (int off = 1; off < 16; off <<= 1)
                psum += __shfl_xor(psum, off, 32);
            l[g] = l[g] * alpha + psum;
#pragma unroll
            for (int n = 0; n < 8; n++) O[n][g] *= alpha;
        }
        asm volatile("s_wait_dscnt 0x0" ::: "memory");  // same-wave P store->load

        // ---- O += P . V ----
        H16 pa[2];
#pragma unroll
        for (int c = 0; c < 2; c++) {
            const _Float16* pp = &Psm[wave][l16][c * 32 + hi * 8];
            pa[c].h[0] = *(const v8h*)pp;
            pa[c].h[1] = *(const v8h*)(pp + 16);
        }
#pragma unroll
        for (int n = 0; n < 8; n++) {
#pragma unroll
            for (int c = 0; c < 2; c++) {
                H16 bf;  // B = V: lane = d col, halves = keys {8hi.., 16+8hi..}
                const _Float16* vp = &Vt[n * 16 + l16][c * 32 + hi * 8];
                bf.h[0] = *(const v8h*)vp;
                bf.h[1] = *(const v8h*)(vp + 16);
                O[n] = wmma_f16(pa[c], bf, O[n]);
            }
        }
        __syncthreads();
    }

    // ---- normalize + write y f16 as [B, T, H, D] (row-major 4096x2048) ----
    const int b = bh >> 4;
    const int h = bh & 15;
#pragma unroll
    for (int g = 0; g < 8; g++) {
        const float inv = 1.0f / l[g];
        const int row = qr0 + g + hi * 8;
#pragma unroll
        for (int n = 0; n < 8; n++) {
            const int d = n * 16 + l16;
            Yh[(((size_t)b * T_ + row) * H_ + h) * D_ + d] = (_Float16)(O[n][g] * inv);
        }
    }
}

// ---------------------------------------------------------------------------
// Launch
// ---------------------------------------------------------------------------
extern "C" void kernel_launch(void* const* d_in, const int* in_sizes, int n_in,
                              void* d_out, int out_size, void* d_ws, size_t ws_size,
                              hipStream_t stream) {
    (void)in_sizes; (void)n_in; (void)out_size; (void)ws_size;
    const float* x      = (const float*)d_in[0];
    const float* w_attn = (const float*)d_in[1];
    const float* b_attn = (const float*)d_in[2];
    const float* w_proj = (const float*)d_in[3];
    const float* b_proj = (const float*)d_in[4];
    float* out = (float*)d_out;

    char* ws = (char*)d_ws;
    size_t off = 0;
    auto take = [&](size_t halves) -> _Float16* {
        _Float16* p = (_Float16*)(ws + off);
        off += (halves * sizeof(_Float16) + 255) & ~(size_t)255;
        return p;
    };
    _Float16* xh  = take((size_t)4096 * 2048);
    _Float16* wah = take((size_t)2048 * 6144);
    _Float16* wph = take((size_t)2048 * 2048);
    _Float16* qh  = take((size_t)B_ * H_ * T_ * D_);
    _Float16* kh  = take((size_t)B_ * H_ * T_ * D_);
    _Float16* vh  = take((size_t)B_ * H_ * T_ * D_);
    _Float16* yh  = take((size_t)4096 * 2048);

    f32_to_f16_kernel<<<2048, 256, 0, stream>>>(x, xh, (size_t)4096 * 2048);
    f32_to_f16_kernel<<<2048, 256, 0, stream>>>(w_attn, wah, (size_t)2048 * 6144);
    f32_to_f16_kernel<<<2048, 256, 0, stream>>>(w_proj, wph, (size_t)2048 * 2048);

    // QKV: [4096,2048] x [2048,6144]
    gemm_kernel<1><<<dim3(48, 32), 256, 0, stream>>>(
        xh, wah, b_attn, nullptr, qh, kh, vh, 4096, 6144, 2048);

    const float qscale = 0.08838834764831845f;  // 1/sqrt(128)
    rope_kernel<<<16384, 256, 0, stream>>>(qh, qscale);
    rope_kernel<<<16384, 256, 0, stream>>>(kh, 1.0f);

    flash_kernel<<<512, 256, 0, stream>>>(qh, kh, vh, yh);

    // proj: [4096,2048] x [2048,2048] -> f32 out
    gemm_kernel<0><<<dim3(16, 32), 256, 0, stream>>>(
        yh, wph, b_proj, out, nullptr, nullptr, nullptr, 4096, 2048, 2048);
}
